// DebertaV2Attention_63977832841385
// MI455X (gfx1250) — matrix operand order repaired and checked by reading, not compiled, not run on hardware
//
#include <hip/hip_runtime.h>
#include <hip/hip_bf16.h>

// ---------------------------------------------------------------------------
// DeBERTa-v2 disentangled attention forward, CDNA5 (gfx1250), wave32 + WMMA.
// B=4, N=1024, H=16, D=64, HID=1024, ATT_SPAN=256 (2*SPAN=512 buckets).
// ---------------------------------------------------------------------------

typedef __attribute__((ext_vector_type(16))) __bf16 v16bf;
typedef __attribute__((ext_vector_type(8)))  float  v8f;
typedef __attribute__((ext_vector_type(4)))  __bf16 bf16x4;

#define H_    16
#define D_    64
#define N_    1024
#define B_    4
#define HID_  1024
#define SPAN2 512
#define NEG_INF -3.0e38f

__device__ inline v8f zero8() {
  v8f z;
  #pragma unroll
  for (int i = 0; i < 8; ++i) z[i] = 0.f;
  return z;
}

union FragCvt { struct { uint4 lo, hi; } q; v16bf v; };
__device__ inline v16bf mkfrag(uint4 lo, uint4 hi) {
  FragCvt f; f.q.lo = lo; f.q.hi = hi; return f.v;
}

__device__ inline v8f wmma_bf16(v16bf a, v16bf b, v8f c) {
  // D = A(16x32 bf16) x B(32x16 bf16) + C(16x16 f32)
  return __builtin_amdgcn_wmma_f32_16x16x32_bf16(
      /*neg_a=*/false, a, /*neg_b=*/false, b,
      /*c_mod=*/(short)0, c, /*reuse_a=*/false, /*reuse_b=*/false);
}

// Low 32 bits of a generic shared-aperture address are the LDS byte offset
// async-to-LDS takes that offset per lane.
__device__ inline unsigned lds_off(const void* p) {
  return (unsigned)(unsigned long long)p;
}

// ---------------------------------------------------------------------------
// Kernel 0: relative-position bucket index table.
// idx(delta) = clip(bucket(delta) + 256, 0, 511), delta in [-1023, 1023].
// bucket is odd => both c2p and p2c gathers use idx(q-k).
// ---------------------------------------------------------------------------
__global__ void build_idx_kernel(int* __restrict__ tab) {
  int i = blockIdx.x * blockDim.x + threadIdx.x;
  if (i >= 2 * N_ - 1) return;
  int delta = i - (N_ - 1);
  const int mid = 128;  // ATT_SPAN/2
  double bucket;
  if (delta >= -mid && delta <= mid) {
    bucket = (double)delta;
  } else {
    double ap = fabs((double)delta);
    double lp = ceil(log(ap / (double)mid) / log(511.0 / (double)mid) *
                     (double)(mid - 1)) + (double)mid;
    bucket = (delta < 0) ? -lp : lp;
  }
  int idx = (int)bucket + 256;
  idx = idx < 0 ? 0 : (idx > SPAN2 - 1 ? SPAN2 - 1 : idx);
  tab[i] = idx;
}

// ---------------------------------------------------------------------------
// Generic bf16 WMMA GEMM: C[M,N] = A[M,K] * B[N,K]^T (+bias/residual).
// Block tile 64x64, K-step 32, 4 waves (128 threads), wave owns 16 M rows.
// Sources converted to bf16 while staging into LDS (vectorized 16B loads).
// MODE: 0 bf16 QKV [B,H,N,D] | 1 bf16 pos [H,512,D] | 2 bf16 batched plain
//       3 f32 + bias + residual
// All per-lane offsets are 32-bit so loads use saddr + voffset addressing.
// ---------------------------------------------------------------------------
#define GT_K  32
#define LDA_S 40   // 32 + 8 pad, rows stay 16B aligned (80B)
#define LDB_S 72   // 64 + 8 pad (144B)

template <int MODE, bool A_BF16, bool B_BF16, bool HAS_BIAS, bool HAS_RES>
__global__ __launch_bounds__(128)
void gemm_wmma_kernel(const void* __restrict__ Ap, int lda, long aBS,
                      const void* __restrict__ Bp, int ldb, long bBS, int bMod,
                      const float* __restrict__ bias,
                      const float* __restrict__ residual,
                      void* __restrict__ Cout, long cBS, int N, int K) {
  __shared__ __align__(16) __bf16 sA[64 * LDA_S];
  __shared__ __align__(16) __bf16 sB[GT_K * LDB_S];

  const int tid = threadIdx.x, wave = tid >> 5, lane = tid & 31;
  const int tm = blockIdx.y * 64, tn = blockIdx.x * 64;
  const int z = blockIdx.z;

  // uniform 64-bit batch bases, computed once
  const float*  A32 = (const float*)Ap  + (A_BF16 ? 0 : (long)z * aBS);
  const __bf16* A16 = (const __bf16*)Ap + (A_BF16 ? (long)z * aBS : 0);
  const float*  B32 = (const float*)Bp  + (B_BF16 ? 0 : (long)(z % bMod) * bBS);
  const __bf16* B16 = (const __bf16*)Bp + (B_BF16 ? (long)(z % bMod) * bBS : 0);

  v8f acc[4];
  #pragma unroll
  for (int i = 0; i < 4; ++i) acc[i] = zero8();

  for (int k0 = 0; k0 < K; k0 += GT_K) {
    // ---- stage A tile [64 rows][32 k] ----
    if (A_BF16) {
      #pragma unroll
      for (int p = 0; p < 2; ++p) {           // 256 x uint4 (8 bf16)
        int g = p * 128 + tid;
        int r = g >> 2, c = (g & 3) * 8;
        *(uint4*)&sA[r * LDA_S + c] =
            *(const uint4*)&A16[(unsigned)((tm + r) * lda + k0 + c)];
      }
    } else {
      #pragma unroll
      for (int p = 0; p < 4; ++p) {           // 512 x float4
        int g = p * 128 + tid;
        int r = g >> 3, c = (g & 7) * 4;
        const float4 f = *(const float4*)&A32[(unsigned)((tm + r) * lda + k0 + c)];
        bf16x4 o = { (__bf16)f.x, (__bf16)f.y, (__bf16)f.z, (__bf16)f.w };
        *(bf16x4*)&sA[r * LDA_S + c] = o;
      }
    }
    // ---- stage B tile [64 n][32 k], transposed -> LDS [32 k][64 n] ----
    if (B_BF16) {
      #pragma unroll
      for (int p = 0; p < 2; ++p) {
        int g = p * 128 + tid;
        int n = g >> 2, c = (g & 3) * 8;
        uint4 u = *(const uint4*)&B16[(unsigned)((tn + n) * ldb + k0 + c)];
        const __bf16* e = (const __bf16*)&u;
        #pragma unroll
        for (int j = 0; j < 8; ++j) sB[(c + j) * LDB_S + n] = e[j];
      }
    } else {
      #pragma unroll
      for (int p = 0; p < 4; ++p) {
        int g = p * 128 + tid;
        int n = g >> 3, c = (g & 7) * 4;
        const float4 f = *(const float4*)&B32[(unsigned)((tn + n) * ldb + k0 + c)];
        sB[(c + 0) * LDB_S + n] = (__bf16)f.x;
        sB[(c + 1) * LDB_S + n] = (__bf16)f.y;
        sB[(c + 2) * LDB_S + n] = (__bf16)f.z;
        sB[(c + 3) * LDB_S + n] = (__bf16)f.w;
      }
    }
    __syncthreads();

    // A fragment: lane<16 holds K {0..7,16..23}, lane>=16 holds {8..15,24..31}
    const int arow = wave * 16 + (lane & 15);
    const int ksel = (lane >> 4) << 3;
    v16bf af = mkfrag(*(const uint4*)&sA[arow * LDA_S + ksel],
                      *(const uint4*)&sA[arow * LDA_S + 16 + ksel]);
    #pragma unroll
    for (int nt = 0; nt < 4; ++nt) {
      // B fragment: N col = lane&15, K {0..7,16..23} (lane<16) / {8..15,24..31}
      const __bf16* bp = &sB[nt * 16 + (lane & 15)];
      v16bf bf;
      #pragma unroll
      for (int s = 0; s < 16; ++s) bf[s] = bp[(ksel + (s & 7) + ((s >> 3) << 4)) * LDB_S];
      acc[nt] = wmma_bf16(af, bf, acc[nt]);
    }
    __syncthreads();
  }

  // epilogue (C/D layout: lanes 0-15 -> M=r, lanes 16-31 -> M=r+8, N=lane&15)
  __bf16* C16z = (__bf16*)Cout + (MODE == 2 ? (long)z * cBS : 0);
  #pragma unroll
  for (int nt = 0; nt < 4; ++nt) {
    #pragma unroll
    for (int r = 0; r < 8; ++r) {
      int row = tm + wave * 16 + r + ((lane >> 4) << 3);
      int col = tn + nt * 16 + (lane & 15);
      float v = acc[nt][r];
      if (HAS_BIAS) v += bias[(unsigned)col];
      if (MODE == 0) {        // QKV: [B,H,N,D]
        unsigned dst = (unsigned)((((row >> 10) * H_ + (col >> 6)) * N_ +
                                   (row & (N_ - 1))) * D_ + (col & (D_ - 1)));
        C16z[dst] = (__bf16)v;
      } else if (MODE == 1) { // positional: [H,512,D]
        unsigned dst = (unsigned)(((col >> 6) * SPAN2 + row) * D_ + (col & (D_ - 1)));
        C16z[dst] = (__bf16)v;
      } else if (MODE == 2) { // batched plain bf16
        C16z[(unsigned)(row * N + col)] = (__bf16)v;
      } else {                // f32 + residual
        float* C = (float*)Cout;
        unsigned dst = (unsigned)(row * N + col);
        float res = HAS_RES ? residual[dst] : 0.f;
        C[dst] = v + res;
      }
    }
  }
}

// ---------------------------------------------------------------------------
// Flash-style fused attention: per (b,h,64-row q block).
// scores = (q.k^T + c2p[q,idx(q-k)] + p2c[k,idx(q-k)]) * invScale, masked,
// online softmax, ctx += P x V via WMMA. k-tiles of 32 so P x V is K=32.
// V tile staged through the CDNA5 async-to-LDS path (ASYNCcnt).
// ---------------------------------------------------------------------------
__global__ __launch_bounds__(128)
void flash_attn_kernel(const __bf16* __restrict__ q, const __bf16* __restrict__ k,
                       const __bf16* __restrict__ v,
                       const __bf16* __restrict__ c2p, const __bf16* __restrict__ p2c,
                       const unsigned char* __restrict__ mask,
                       const int* __restrict__ idxTab,
                       __bf16* __restrict__ ctx, float invScale) {
  const int qblk = blockIdx.x, h = blockIdx.y, b = blockIdx.z;
  const int tid = threadIdx.x, wave = tid >> 5, lane = tid & 31;
  const long bh = (long)b * H_ + h;
  // uniform 64-bit bases; all per-lane offsets below stay unsigned 32-bit
  const __bf16* qh = q + bh * (long)(N_ * D_);
  const __bf16* kh = k + bh * (long)(N_ * D_);
  const __bf16* vh = v + bh * (long)(N_ * D_);
  const __bf16* c2ph = c2p + bh * (long)(N_ * SPAN2);
  const __bf16* p2ch = p2c + bh * (long)(N_ * SPAN2);
  const unsigned char* mk = mask + (long)b * (N_ * N_);

  __shared__ __align__(16) __bf16 sK[64 * 40];     // [d][kcol], ld=40
  __shared__ __align__(16) __bf16 sV[32 * 72];     // [k][d],    ld=72
  __shared__ __align__(16) __bf16 sP[4][16 * 40];  // per-wave P scratch

  const int qBase = qblk * 64 + wave * 16;
  const int mofs = (lane >> 4) << 3;   // row offset for C-layout (0 or 8)
  const int ksel = (lane >> 4) << 3;   // A-layout K selector

  // q A-fragments for K=0..31 and 32..63, loaded straight from global
  v16bf aq[2];
  {
    const __bf16* qp = qh + (unsigned)((qBase + (lane & 15)) * D_);
    aq[0] = mkfrag(*(const uint4*)&qp[ksel],      *(const uint4*)&qp[16 + ksel]);
    aq[1] = mkfrag(*(const uint4*)&qp[32 + ksel], *(const uint4*)&qp[48 + ksel]);
  }

  v8f accO[4];
  #pragma unroll
  for (int i = 0; i < 4; ++i) accO[i] = zero8();
  float mrow[8], lrow[8];
  #pragma unroll
  for (int r = 0; r < 8; ++r) { mrow[r] = NEG_INF; lrow[r] = 0.f; }

  for (int kt = 0; kt < N_ / 32; ++kt) {
    const int kBase = kt * 32;
    // ---- V tile via async copy global->LDS (16B per lane per op) ----
    #pragma unroll
    for (int p = 0; p < 2; ++p) {
      int g = p * 128 + tid;                 // 256 groups of 8 bf16
      int c = g >> 3, d0 = (g & 7) * 8;
      unsigned dst = lds_off(&sV[c * 72 + d0]);
      const __bf16* src = &vh[(unsigned)((kBase + c) * D_ + d0)];
      asm volatile("global_load_async_to_lds_b128 %0, %1, off"
                   :: "v"(dst), "v"(src) : "memory");
    }
    // ---- K tile, transposed [d][kcol]; global side vectorized ----
    #pragma unroll
    for (int p = 0; p < 2; ++p) {
      int g = p * 128 + tid;                 // 32 c-rows x 8 d-groups
      int c = g >> 3, d0 = (g & 7) * 8;
      uint4 u = *(const uint4*)&kh[(unsigned)((kBase + c) * D_ + d0)];
      const __bf16* e = (const __bf16*)&u;
      #pragma unroll
      for (int j = 0; j < 8; ++j) sK[(d0 + j) * 40 + c] = e[j];
    }
    asm volatile("s_wait_asynccnt 0x0" ::: "memory");
    __syncthreads();

    // scores: 16q x 32k in two C fragments
    v8f S[2]; S[0] = zero8(); S[1] = zero8();
    #pragma unroll
    for (int ks = 0; ks < 2; ++ks) {
      #pragma unroll
      for (int nt = 0; nt < 2; ++nt) {
        const __bf16* bp = &sK[(ks * 32) * 40 + nt * 16 + (lane & 15)];
        v16bf bf;
        #pragma unroll
        for (int s = 0; s < 16; ++s) bf[s] = bp[(ksel + (s & 7) + ((s >> 3) << 4)) * 40];
        S[nt] = wmma_bf16(aq[ks], bf, S[nt]);
      }
    }

    // disentangled bias gathers + scale + mask
    // (unsigned 32-bit offsets => saddr + voffset + scale_offset loads)
    #pragma unroll
    for (int nt = 0; nt < 2; ++nt) {
      int kg = kBase + nt * 16 + (lane & 15);
      #pragma unroll
      for (int r = 0; r < 8; ++r) {
        int qg = qBase + r + mofs;
        float s = S[nt][r];
        unsigned idx = (unsigned)idxTab[(unsigned)(qg - kg + (N_ - 1))];
        s += (float)c2ph[(unsigned)(qg * SPAN2) + idx] +
             (float)p2ch[(unsigned)(kg * SPAN2) + idx];
        s *= invScale;
        if (!mk[(unsigned)(qg * N_ + kg)]) s = NEG_INF;
        S[nt][r] = s;
      }
    }

    // online softmax (row reductions across 16 lanes per half-wave)
    float alpha[8];
    #pragma unroll
    for (int r = 0; r < 8; ++r) {
      float mx = fmaxf(S[0][r], S[1][r]);
      #pragma unroll
      for (int o = 1; o < 16; o <<= 1) mx = fmaxf(mx, __shfl_xor(mx, o, 32));
      float mn = fmaxf(mrow[r], mx);
      alpha[r] = __expf(mrow[r] - mn);
      mrow[r] = mn;
      float e0 = (S[0][r] <= -1.0e37f) ? 0.f : __expf(S[0][r] - mn);
      float e1 = (S[1][r] <= -1.0e37f) ? 0.f : __expf(S[1][r] - mn);
      S[0][r] = e0; S[1][r] = e1;
      float ps = e0 + e1;
      #pragma unroll
      for (int o = 1; o < 16; o <<= 1) ps += __shfl_xor(ps, o, 32);
      lrow[r] = lrow[r] * alpha[r] + ps;
    }
    #pragma unroll
    for (int dt = 0; dt < 4; ++dt) {
      #pragma unroll
      for (int r = 0; r < 8; ++r) accO[dt][r] *= alpha[r];
    }

    // P: C-layout -> LDS (wave-private) -> A-layout fragment
    __bf16* sp = sP[wave];
    #pragma unroll
    for (int nt = 0; nt < 2; ++nt) {
      #pragma unroll
      for (int r = 0; r < 8; ++r)
        sp[(r + mofs) * 40 + nt * 16 + (lane & 15)] = (__bf16)S[nt][r];
    }
    v16bf pf = mkfrag(*(const uint4*)&sp[(lane & 15) * 40 + ksel],
                      *(const uint4*)&sp[(lane & 15) * 40 + 16 + ksel]);

    // ctx accumulation: O(16x64) += P(16x32) x V(32x64)
    #pragma unroll
    for (int dt = 0; dt < 4; ++dt) {
      const __bf16* vp = &sV[dt * 16 + (lane & 15)];
      v16bf bf;
      #pragma unroll
      for (int s = 0; s < 16; ++s) bf[s] = vp[(ksel + (s & 7) + ((s >> 3) << 4)) * 72];
      accO[dt] = wmma_bf16(pf, bf, accO[dt]);
    }
    __syncthreads();
  }

  // normalize and store ctx as [B,N,HID] bf16 (heads re-interleaved)
  float inv[8];
  #pragma unroll
  for (int r = 0; r < 8; ++r) inv[r] = lrow[r] > 0.f ? 1.f / lrow[r] : 0.f;
  #pragma unroll
  for (int dt = 0; dt < 4; ++dt) {
    #pragma unroll
    for (int r = 0; r < 8; ++r) {
      int qg = qBase + r + mofs;
      int dc = dt * 16 + (lane & 15);
      ctx[(unsigned)((b * N_ + qg) * HID_ + h * D_ + dc)] =
          (__bf16)(accO[dt][r] * inv[r]);
    }
  }
}

// ---------------------------------------------------------------------------
// LayerNorm over HID=1024, one row per block.
// ---------------------------------------------------------------------------
__global__ __launch_bounds__(256)
void layernorm_kernel(const float* __restrict__ x, const float* __restrict__ g,
                      const float* __restrict__ be, float* __restrict__ out) {
  __shared__ float red[256];
  const int row = blockIdx.x, tid = threadIdx.x;
  const float* xr = x + (long)row * HID_;
  float s = 0.f;
  for (int i = tid; i < HID_; i += 256) s += xr[i];
  red[tid] = s; __syncthreads();
  for (int o = 128; o > 0; o >>= 1) { if (tid < o) red[tid] += red[tid + o]; __syncthreads(); }
  float mu = red[0] / HID_; __syncthreads();
  float vv = 0.f;
  for (int i = tid; i < HID_; i += 256) { float d = xr[i] - mu; vv += d * d; }
  red[tid] = vv; __syncthreads();
  for (int o = 128; o > 0; o >>= 1) { if (tid < o) red[tid] += red[tid + o]; __syncthreads(); }
  float inv = rsqrtf(red[0] / HID_ + 1e-7f);
  for (int i = tid; i < HID_; i += 256)
    out[(long)row * HID_ + i] = (xr[i] - mu) * inv * g[i] + be[i];
}

// ---------------------------------------------------------------------------
extern "C" void kernel_launch(void* const* d_in, const int* in_sizes, int n_in,
                              void* d_out, int out_size, void* d_ws, size_t ws_size,
                              hipStream_t stream) {
  (void)in_sizes; (void)n_in; (void)out_size; (void)ws_size;
  const float* hs  = (const float*)d_in[0];
  const unsigned char* mask = (const unsigned char*)d_in[1];
  const float* rel = (const float*)d_in[2];
  const float* Wq = (const float*)d_in[3];  const float* bq = (const float*)d_in[4];
  const float* Wk = (const float*)d_in[5];  const float* bk = (const float*)d_in[6];
  const float* Wv = (const float*)d_in[7];  const float* bv = (const float*)d_in[8];
  const float* Wo = (const float*)d_in[9];  const float* bo = (const float*)d_in[10];
  const float* lng = (const float*)d_in[11];
  const float* lnb = (const float*)d_in[12];
  float* out = (float*)d_out;

  // workspace carve-out
  char* ws = (char*)d_ws; size_t off = 0;
  auto alloc = [&](size_t bytes) -> void* {
    off = (off + 255) & ~(size_t)255; void* p = ws + off; off += bytes; return p;
  };
  const size_t szQKV = (size_t)B_ * H_ * N_ * D_ * 2;      // 8 MB each
  const size_t szPos = (size_t)H_ * SPAN2 * D_ * 2;        // 1 MB each
  const size_t szRel = (size_t)B_ * H_ * N_ * SPAN2 * 2;   // 64 MB each
  const size_t szCtx = (size_t)B_ * N_ * HID_ * 2;         // 8 MB
  const size_t szTmp = (size_t)B_ * N_ * HID_ * 4;         // 16 MB

  int*     idxTab = (int*)alloc((2 * N_ - 1) * sizeof(int));
  __bf16*  qb   = (__bf16*)alloc(szQKV);
  __bf16*  kb   = (__bf16*)alloc(szQKV);
  __bf16*  vb   = (__bf16*)alloc(szQKV);
  __bf16*  posk = (__bf16*)alloc(szPos);
  __bf16*  posq = (__bf16*)alloc(szPos);
  __bf16*  c2pb = (__bf16*)alloc(szRel);
  __bf16*  p2cb = (__bf16*)alloc(szRel);
  __bf16*  ctxb = (__bf16*)alloc(szCtx);
  float*   tmpf = (float*)alloc(szTmp);

  const float invScale = 0.07216878364870323f;  // 1/sqrt(64*3)
  dim3 blk(128);

  // 0) relative-position index table
  build_idx_kernel<<<dim3((2 * N_ - 1 + 255) / 256), dim3(256), 0, stream>>>(idxTab);

  // 1) Q/K/V projections: [4096,1024] x [1024,1024]^T -> bf16 [B,H,N,D]
  dim3 gProj(HID_ / 64, (B_ * N_) / 64, 1);
  gemm_wmma_kernel<0, false, false, true, false><<<gProj, blk, 0, stream>>>(
      hs, HID_, 0, Wq, HID_, 0, 1, bq, nullptr, qb, 0, HID_, HID_);
  gemm_wmma_kernel<0, false, false, true, false><<<gProj, blk, 0, stream>>>(
      hs, HID_, 0, Wk, HID_, 0, 1, bk, nullptr, kb, 0, HID_, HID_);
  gemm_wmma_kernel<0, false, false, true, false><<<gProj, blk, 0, stream>>>(
      hs, HID_, 0, Wv, HID_, 0, 1, bv, nullptr, vb, 0, HID_, HID_);

  // 2) positional projections: [512,1024] x W^T -> bf16 [H,512,D]
  dim3 gPos(HID_ / 64, SPAN2 / 64, 1);
  gemm_wmma_kernel<1, false, false, true, false><<<gPos, blk, 0, stream>>>(
      rel, HID_, 0, Wk, HID_, 0, 1, bk, nullptr, posk, 0, HID_, HID_);
  gemm_wmma_kernel<1, false, false, true, false><<<gPos, blk, 0, stream>>>(
      rel, HID_, 0, Wq, HID_, 0, 1, bq, nullptr, posq, 0, HID_, HID_);

  // 3) c2p = q x pos_k^T, p2c = k x pos_q^T : batched over B*H, [1024,512,64]
  dim3 gRel(SPAN2 / 64, N_ / 64, B_ * H_);
  gemm_wmma_kernel<2, true, true, false, false><<<gRel, blk, 0, stream>>>(
      qb, D_, (long)N_ * D_, posk, D_, (long)SPAN2 * D_, H_,
      nullptr, nullptr, c2pb, (long)N_ * SPAN2, SPAN2, D_);
  gemm_wmma_kernel<2, true, true, false, false><<<gRel, blk, 0, stream>>>(
      kb, D_, (long)N_ * D_, posq, D_, (long)SPAN2 * D_, H_,
      nullptr, nullptr, p2cb, (long)N_ * SPAN2, SPAN2, D_);

  // 4) fused disentangled attention (scores + softmax + ctx)
  dim3 gFl(N_ / 64, H_, B_);
  flash_attn_kernel<<<gFl, blk, 0, stream>>>(qb, kb, vb, c2pb, p2cb, mask, idxTab,
                                             ctxb, invScale);

  // 5) output projection + bias + residual -> f32
  gemm_wmma_kernel<3, true, false, true, true><<<gProj, blk, 0, stream>>>(
      ctxb, HID_, 0, Wo, HID_, 0, 1, bo, hs, tmpf, 0, HID_, HID_);

  // 6) LayerNorm
  layernorm_kernel<<<dim3(B_ * N_), dim3(256), 0, stream>>>(tmpf, lng, lnb, out);
}
